// VectorQuantizerEMA_43679817401104
// MI455X (gfx1250) — matrix operand order repaired and checked by reading, not compiled, not run on hardware
//
#include <hip/hip_runtime.h>
#include <hip/hip_bf16.h>

// ---------------------------------------------------------------------------
// VQ-VAE EMA vector quantizer for MI455X (gfx1250, wave32, WMMA)
//   z_e:        (16, 256, 2048) f32   -> N = 32768 rows, D = 256
//   embedding:  (8192, 256) f32       -> K = 8192 codes
// Distance GEMM (N x K x D, 137 GFLOP) fused with argmin:
//   v_wmma_f32_16x16x32_bf16; 32-code B tiles double-buffered in LDS via
//   global_load_async_to_lds_b128 (ASYNCcnt); all 8 B fragments preloaded
//   per sub-tile so WMMAs issue back-to-back.
// ---------------------------------------------------------------------------

typedef __attribute__((ext_vector_type(16))) __bf16 v16bf;
typedef __attribute__((ext_vector_type(8)))  __bf16 v8bf;
typedef __attribute__((ext_vector_type(8)))  float  v8f;
typedef int v4i __attribute__((vector_size(16)));   // b128 payload type

#define AS1 __attribute__((address_space(1)))
#define AS3 __attribute__((address_space(3)))

#if defined(__has_builtin)
#if __has_builtin(__builtin_amdgcn_global_load_async_to_lds_b128)
#define HAVE_ASYNC_LDS 1
#endif
#endif

namespace vq {
constexpr int KC   = 8192;   // num codes
constexpr int D    = 256;    // code dim
constexpr int BB   = 16;     // batch
constexpr int T    = 2048;   // time
constexpr int N    = BB * T; // 32768 rows
constexpr float DECAY    = 0.99f;
constexpr float EPS      = 1e-5f;
constexpr float COMMIT_W = 1.0f;
// LDS tile rows padded 512B -> 528B (132 dwords, 132 % 64 == 4 -> lane start
// banks 4*row16: conflict-free b128 reads).
constexpr int ROWB  = 528;
constexpr int CHUNK = 32;            // codes per barrier iteration
constexpr int NSUB  = CHUNK / 16;    // 16x16 WMMA sub-tiles per chunk
}

__device__ __forceinline__ void wait_async_zero() {
#if defined(HAVE_ASYNC_LDS)
#if __has_builtin(__builtin_amdgcn_s_wait_asynccnt)
    __builtin_amdgcn_s_wait_asynccnt(0);
#else
    asm volatile("s_wait_asynccnt 0x0" ::: "memory");
#endif
#endif
}

// Cooperative copy of one CHUNK x 256 bf16 code tile into an LDS buffer.
// CHUNK*32 x 16B transfers; 256 threads x CHUNK/8. Row r -> offset r*ROWB.
__device__ __forceinline__ void stage_tile(const __bf16* __restrict__ eb,
                                           int cbase, char* lbuf, int tid) {
    using namespace vq;
    #pragma unroll
    for (int i = 0; i < CHUNK / 8; ++i) {
        const int c   = tid + i * 256;   // 0 .. CHUNK*32-1
        const int row = c >> 5;          // 0 .. CHUNK-1
        const int col = c & 31;          // 0..31 (16B units)
        const __bf16* gsrc = eb + (size_t)(cbase + row) * D + col * 8;
        char* dst = lbuf + row * ROWB + col * 16;
#if defined(HAVE_ASYNC_LDS)
        __builtin_amdgcn_global_load_async_to_lds_b128(
            (AS1 v4i*)(AS1 void*)(void*)const_cast<__bf16*>(gsrc),
            (AS3 v4i*)(AS3 void*)dst, 0, 0);
#else
        *(uint4*)dst = *(const uint4*)gsrc;   // sync fallback (regs + ds_store)
#endif
    }
}

// --------------------------- codebook prep ---------------------------------
__global__ __launch_bounds__(256)
void prep_codes(const float* __restrict__ emb, __bf16* __restrict__ eb,
                float* __restrict__ enorm) {
    using namespace vq;
    const int k = blockIdx.x;
    const int d = threadIdx.x;
    const float e = emb[(size_t)k * D + d];
    eb[(size_t)k * D + d] = (__bf16)e;

    float sq = e * e;
    #pragma unroll
    for (int m = 16; m >= 1; m >>= 1) sq += __shfl_xor(sq, m, 32);
    __shared__ float part[8];
    if ((d & 31) == 0) part[d >> 5] = sq;
    __syncthreads();
    if (d == 0) {
        float s = 0.f;
        #pragma unroll
        for (int i = 0; i < 8; ++i) s += part[i];
        enorm[k] = s;
    }
}

// ----------------------------- z prep --------------------------------------
// Transpose-convert z_e (B, D, T) f32 -> zb (N, D) bf16 via a 32x32 LDS tile.
__global__ __launch_bounds__(256)
void prep_z(const float* __restrict__ z, __bf16* __restrict__ zb) {
    using namespace vq;
    __shared__ float tile[32][33];
    const int n0 = blockIdx.x * 32;
    const int d0 = blockIdx.y * 32;
    const int b  = n0 / T;
    const int t0 = n0 % T;
    const int tx = threadIdx.x;
    const int ty = threadIdx.y;

    const float* base = z + (size_t)b * D * T + (size_t)d0 * T + t0;
    #pragma unroll
    for (int r = 0; r < 4; ++r) {
        const int d = ty + 8 * r;
        tile[d][tx] = base[(size_t)d * T + tx];
    }
    __syncthreads();
    #pragma unroll
    for (int r = 0; r < 4; ++r) {
        const int row = ty + 8 * r;
        zb[(size_t)(n0 + row) * D + d0 + tx] = (__bf16)tile[tx][row];
    }
}

// ------------------------ fused GEMM + argmin -------------------------------
// 256 threads = 8 waves; each wave owns 16 rows (A register-resident for the
// whole D=256). Per 32-code chunk: next tile staged async into LDS while the
// current tile feeds 2 x (16 ds_load_b128 clause + 8 WMMAs); argmin fused.
__global__ __launch_bounds__(256)
void vq_argmin(const __bf16* __restrict__ zb, const __bf16* __restrict__ eb,
               const float* __restrict__ enorm, int* __restrict__ idx_out) {
    using namespace vq;
    __shared__ __align__(16) char lbs[2][CHUNK * ROWB];

    const int tid   = threadIdx.x;
    const int lane  = tid & 31;
    const int wave  = tid >> 5;
    const int row16 = lane & 15;
    const int half  = lane >> 4;
    const int rowbase = blockIdx.x * 128 + wave * 16;

    // A fragments: lane holds row (rowbase+row16); step j covers d run
    // [d0+16*half .. +15].
    v16bf a[8];
    const __bf16* zrow = zb + (size_t)(rowbase + row16) * D;
    #pragma unroll
    for (int j = 0; j < 8; ++j) {
        const int d0 = j * 32;
        a[j] = *(const v16bf*)(zrow + d0 + 16 * half);
    }

    float bestv[8];
    int   besti[8];
    #pragma unroll
    for (int v = 0; v < 8; ++v) { bestv[v] = 3.4e38f; besti[v] = 0x7fffffff; }

    // Prologue: stage chunk 0, wait, barrier.
    stage_tile(eb, 0, lbs[0], tid);
    wait_async_zero();
    __syncthreads();

    for (int cbase = 0; cbase < KC; cbase += CHUNK) {
        const int buf = (cbase / CHUNK) & 1;
        char* cur = lbs[buf];

        // Kick off async copy of the next tile into the other buffer.
        if (cbase + CHUNK < KC)              // uniform branch
            stage_tile(eb, cbase + CHUNK, lbs[buf ^ 1], tid);

        #pragma unroll
        for (int s = 0; s < NSUB; ++s) {
            // B fragments: lane = code (s*16+row16), d = j*32 + 16*half.
            const char* bb = cur + (s * 16 + row16) * ROWB + half * 32;
            v16bf bf[8];
            #pragma unroll
            for (int j = 0; j < 8; ++j)          // clause of 16 ds_load_b128
                bf[j] = *(const v16bf*)(bb + j * 64);

            v8f acc = {};
            #pragma unroll
            for (int j = 0; j < 8; ++j)          // dense WMMA issue
                acc = __builtin_amdgcn_wmma_f32_16x16x32_bf16(
                    false, a[j], false, bf[j], (short)0, acc, false, false);

            const int   code = cbase + s * 16 + row16;
            const float en   = enorm[code];
            #pragma unroll
            for (int v = 0; v < 8; ++v) {
                const float dist = en - 2.0f * acc[v];   // ||e||^2 - 2 z.e
                if (dist < bestv[v] || (dist == bestv[v] && code < besti[v])) {
                    bestv[v] = dist; besti[v] = code;
                }
            }
        }

        // My async copies done + all waves finished reading `cur`.
        wait_async_zero();
        __syncthreads();
    }

    // Cross-lane argmin within each 16-lane half; ties -> lowest index.
    #pragma unroll
    for (int v = 0; v < 8; ++v) {
        float bv = bestv[v];
        int   bi = besti[v];
        #pragma unroll
        for (int m = 8; m >= 1; m >>= 1) {
            const float ov = __shfl_xor(bv, m, 32);
            const int   oi = __shfl_xor(bi, m, 32);
            if (ov < bv || (ov == bv && oi < bi)) { bv = ov; bi = oi; }
        }
        if (row16 == 0) idx_out[rowbase + v + 8 * half] = bi;
    }
}

// ------------------- gather + segment sums + loss ---------------------------
__global__ __launch_bounds__(256)
void gather_stats(const float* __restrict__ z, const float* __restrict__ emb,
                  const int* __restrict__ idx, float* __restrict__ zq_out,
                  float* __restrict__ idx_out_f, float* __restrict__ sum_counts,
                  float* __restrict__ sum_emb, float* __restrict__ scalars) {
    using namespace vq;
    const int n = blockIdx.x;
    const int d = threadIdx.x;
    const int b = n / T, t = n % T;
    const int k = idx[n];

    const size_t zoff = (size_t)b * D * T + (size_t)d * T + t;
    const float zv = z[zoff];
    const float ev = emb[(size_t)k * D + d];
    zq_out[zoff] = ev;                               // straight-through value
    atomicAdd(sum_emb + (size_t)k * D + d, zv);      // segment_sum(z)

    const float diff = zv - ev;
    float sq = diff * diff;
    #pragma unroll
    for (int m = 16; m >= 1; m >>= 1) sq += __shfl_xor(sq, m, 32);
    __shared__ float part[8];
    if ((d & 31) == 0) part[d >> 5] = sq;
    __syncthreads();
    if (d == 0) {
        float s = 0.f;
        #pragma unroll
        for (int i = 0; i < 8; ++i) s += part[i];
        atomicAdd(&scalars[0], s);                   // sum sq diffs
        atomicAdd(sum_counts + k, 1.0f);             // segment_sum(ones)
        idx_out_f[n] = (float)k;
    }
}

// ------------------- EMA cluster sizes + global sums ------------------------
__global__ __launch_bounds__(256)
void ema_counts(const float* __restrict__ cluster_size,
                const float* __restrict__ sum_counts,
                float* __restrict__ new_cs_out, float* __restrict__ scalars) {
    using namespace vq;
    const int k = blockIdx.x * 256 + threadIdx.x;
    const float sc  = sum_counts[k];
    const float ncs = cluster_size[k] * DECAY + (1.0f - DECAY) * sc;
    new_cs_out[k] = ncs;

    float s1 = ncs;
    float s2 = sc > 0.f ? 1.f : 0.f;
    #pragma unroll
    for (int m = 16; m >= 1; m >>= 1) {
        s1 += __shfl_xor(s1, m, 32);
        s2 += __shfl_xor(s2, m, 32);
    }
    __shared__ float p1[8], p2[8];
    const int d = threadIdx.x;
    if ((d & 31) == 0) { p1[d >> 5] = s1; p2[d >> 5] = s2; }
    __syncthreads();
    if (d == 0) {
        float a = 0.f, bsum = 0.f;
        #pragma unroll
        for (int i = 0; i < 8; ++i) { a += p1[i]; bsum += p2[i]; }
        atomicAdd(&scalars[1], a);      // n = sum(new_cluster_size)
        atomicAdd(&scalars[2], bsum);   // utilization count
    }
}

// --------------------- EMA embed_avg + normalization ------------------------
__global__ __launch_bounds__(256)
void ema_final(const float* __restrict__ embed_avg, const float* __restrict__ sum_emb,
               const float* __restrict__ new_cs, const float* __restrict__ scalars,
               float* __restrict__ new_emb_out, float* __restrict__ new_ea_out,
               float* __restrict__ loss_out, float* __restrict__ util_out) {
    using namespace vq;
    const int k = blockIdx.x;
    const int d = threadIdx.x;
    const size_t o = (size_t)k * D + d;
    const float nea = embed_avg[o] * DECAY + (1.0f - DECAY) * sum_emb[o];
    new_ea_out[o] = nea;
    const float n = scalars[1];
    const float smoothed = (new_cs[k] + EPS) / (n + (float)KC * EPS) * n;
    new_emb_out[o] = nea / smoothed;
    if (k == 0 && d == 0) {
        loss_out[0] = scalars[0] * (COMMIT_W / (float)((size_t)N * D));
        util_out[0] = scalars[2] / (float)KC;
    }
}

// ----------------------------- launcher -------------------------------------
extern "C" void kernel_launch(void* const* d_in, const int* in_sizes, int n_in,
                              void* d_out, int out_size, void* d_ws, size_t ws_size,
                              hipStream_t stream) {
    using namespace vq;
    const float* z_e        = (const float*)d_in[0];
    const float* embedding  = (const float*)d_in[1];
    const float* cluster_sz = (const float*)d_in[2];
    const float* embed_avg  = (const float*)d_in[3];

    char* ws = (char*)d_ws;
    const size_t OFF_ZB    = 0;                               // N*D bf16
    const size_t OFF_EB    = OFF_ZB    + (size_t)N * D * 2;   // K*D bf16
    const size_t OFF_ENORM = OFF_EB    + (size_t)KC * D * 2;  // K f32
    const size_t OFF_IDX   = OFF_ENORM + (size_t)KC * 4;      // N i32
    const size_t OFF_SUMC  = OFF_IDX   + (size_t)N * 4;       // K f32 (zeroed)
    const size_t OFF_SUME  = OFF_SUMC  + (size_t)KC * 4;      // K*D f32 (zeroed)

    __bf16* zb    = (__bf16*)(ws + OFF_ZB);
    __bf16* eb    = (__bf16*)(ws + OFF_EB);
    float*  enorm = (float*)(ws + OFF_ENORM);
    int*    idx   = (int*)(ws + OFF_IDX);
    float*  sumc  = (float*)(ws + OFF_SUMC);
    float*  sume  = (float*)(ws + OFF_SUME);
    float*  scal  = (float*)(ws + OFF_SUME + (size_t)KC * D * 4);

    float* out      = (float*)d_out;
    float* o_zq     = out;                         // B*D*T
    float* o_idx    = o_zq  + (size_t)N * D;       // N
    float* o_loss   = o_idx + (size_t)N;           // 1
    float* o_newemb = o_loss + 1;                  // K*D
    float* o_newcs  = o_newemb + (size_t)KC * D;   // K
    float* o_newea  = o_newcs + (size_t)KC;        // K*D
    float* o_util   = o_newea + (size_t)KC * D;    // 1

    (void)hipMemsetAsync(ws + OFF_SUMC,
                         0, (size_t)KC * 4 + (size_t)KC * D * 4 + 16, stream);

    prep_codes<<<KC, 256, 0, stream>>>(embedding, eb, enorm);
    prep_z<<<dim3(N / 32, D / 32), dim3(32, 8), 0, stream>>>(z_e, zb);
    vq_argmin<<<N / 128, 256, 0, stream>>>(zb, eb, enorm, idx);
    gather_stats<<<N, 256, 0, stream>>>(z_e, embedding, idx, o_zq, o_idx,
                                        sumc, sume, scal);
    ema_counts<<<KC / 256, 256, 0, stream>>>(cluster_sz, sumc, o_newcs, scal);
    ema_final<<<KC, 256, 0, stream>>>(embed_avg, sume, o_newcs, scal,
                                      o_newemb, o_newea, o_loss, o_util);
}
